// GraphAttention_75780402970746
// MI455X (gfx1250) — matrix operand order repaired and checked by reading, not compiled, not run on hardware
//
#include <hip/hip_runtime.h>
#include <hip/hip_bf16.h>

typedef __attribute__((ext_vector_type(16))) _Float16 v16h;
typedef __attribute__((ext_vector_type(8)))  float    v8f;
typedef __attribute__((ext_vector_type(4)))  float    v4f;

#define BB 8
#define NN 2048
#define DD 64
#define NEG_SLOPE 0.2f
// (1/TAU) * log2(e): fold temperature and base-2 exp into one scale
#define C_SCALE 0.72134752044448169136f

// ---------------------------------------------------------------------------
// Stage 1: Wx = x @ W ; e_src = Wx . a_src ; e_dst = Wx . a_dst
// Stores WxT as f16 [B, D, N] (transposed) so stage-2 WMMA B-operand loads are
// contiguous along K. Cost is negligible (134 MFLOP, 4 MB in).
// ---------------------------------------------------------------------------
__global__ __launch_bounds__(256) void gat_stage1(
    const float* __restrict__ x, const float* __restrict__ W,
    const float* __restrict__ a_src, const float* __restrict__ a_dst,
    _Float16* __restrict__ WxT, float* __restrict__ es, float* __restrict__ ed)
{
    __shared__ float Ws[DD * DD];
    __shared__ float as_s[DD], ad_s[DD];
    int tid = threadIdx.x;
    for (int i = tid; i < DD * DD; i += 256) Ws[i] = W[i];
    if (tid < DD) { as_s[tid] = a_src[tid]; ad_s[tid] = a_dst[tid]; }
    __syncthreads();

    int row = blockIdx.x * 256 + tid;          // 0 .. B*N-1
    int b = row / NN, n = row % NN;
    const float* xr = x + (size_t)row * DD;

    v4f acc[16];
#pragma unroll
    for (int o = 0; o < 16; o++) acc[o] = (v4f){0.f, 0.f, 0.f, 0.f};

    for (int i = 0; i < DD; i++) {
        float xi = xr[i];
        const v4f* wrow = (const v4f*)(Ws + i * DD);
#pragma unroll
        for (int o = 0; o < 16; o++) acc[o] += xi * wrow[o];
    }

    float s1 = 0.f, s2 = 0.f;
    _Float16* wt = WxT + (size_t)b * DD * NN + n;   // column n of [D,N]
#pragma unroll
    for (int o = 0; o < 16; o++) {
#pragma unroll
        for (int c = 0; c < 4; c++) {
            float v = acc[o][c];
            int oo = o * 4 + c;
            s1 += v * as_s[oo];
            s2 += v * ad_s[oo];
            wt[(size_t)oo * NN] = (_Float16)v;  // coalesced across threads (n)
        }
    }
    es[row] = s1;
    ed[row] = s2;
}

// ---------------------------------------------------------------------------
// Stage 1b: per-batch global max of e_dst. Because LeakyReLU is monotonic,
// leaky(e_src[n] + gmax_b)/TAU upper-bounds every row's score max, which lets
// stage 2 use a FIXED softmax shift (no online rescale, no shuffles in loop).
// ---------------------------------------------------------------------------
__global__ __launch_bounds__(256) void gat_gmax(
    const float* __restrict__ ed, float* __restrict__ gmax)
{
    int b = blockIdx.x, tid = threadIdx.x;
    const float* p = ed + (size_t)b * NN;
    float m = -3.0e38f;
    for (int i = tid; i < NN; i += 256) m = fmaxf(m, p[i]);
#pragma unroll
    for (int o = 16; o > 0; o >>= 1) m = fmaxf(m, __shfl_xor(m, o, 32));
    __shared__ float sm[8];
    if ((tid & 31) == 0) sm[tid >> 5] = m;
    __syncthreads();
    if (tid == 0) {
        float r = sm[0];
#pragma unroll
        for (int i = 1; i < 8; i++) r = fmaxf(r, sm[i]);
        gmax[b] = r;
    }
}

// ---------------------------------------------------------------------------
// Stage 2: one wave32 per 16-row tile. Fixed-shift softmax over m in
// 32-column tiles; P (f16, WMMA A-layout) x WxT (f16, WMMA B-layout) -> f32 C.
// Reads the 134 MB adjacency exactly once (NT hint). Every exp2 argument is
// <= 0 by construction of the bound, so raw v_exp_f32 is safe (underflow -> 0)
// and p in [0,1] is f16-safe. Adjacency is exactly {0,1}: mask = multiply.
// ---------------------------------------------------------------------------
__global__ __launch_bounds__(32) void gat_stage2(
    const float* __restrict__ A, const _Float16* __restrict__ WxT,
    const float* __restrict__ es_g, const float* __restrict__ ed_g,
    const float* __restrict__ gmax, float* __restrict__ out)
{
    const int lane = threadIdx.x;        // 0..31
    const int r    = lane & 15;          // row within 16-row tile
    const int hi   = lane >> 4;          // 0: cols {0..15}, 1: {16..31}
    const int c0   = hi * 16;

    const int tile = blockIdx.x;         // 0..1023
    const int b    = tile >> 7;          // tile / (N/16)
    const int r0   = (tile & 127) << 4;

    const float     esv  = es_g[b * NN + r0 + r];
    const float*    Arow = A + ((size_t)b * NN + (r0 + r)) * NN;
    const float*    edp  = ed_g + (size_t)b * NN;
    const _Float16* Wb   = WxT + (size_t)b * DD * NN;

    // Row-wise softmax shift (upper bound on the row max, in log2 domain).
    const float eb     = esv + gmax[b];
    const float bound2 = fmaxf(eb, NEG_SLOPE * eb) * C_SCALE;

    v8f acc0 = {}, acc1 = {}, acc2 = {}, acc3 = {};
    float lr[4] = {0.f, 0.f, 0.f, 0.f};   // 4 partial sums -> no serial chain

    for (int m = 0; m < NN; m += 32) {
        __builtin_prefetch(Arow + m + 256, 0, 0);   // speculative; OOB dropped

        // Adjacency tile, per-lane columns c0..c0+15 (k = 16*hi + j, same order as B).
        v4f a0 = __builtin_nontemporal_load((const v4f*)(Arow + m + c0));
        v4f a1 = __builtin_nontemporal_load((const v4f*)(Arow + m + c0 + 4));
        v4f a2 = __builtin_nontemporal_load((const v4f*)(Arow + m + c0 + 8));
        v4f a3 = __builtin_nontemporal_load((const v4f*)(Arow + m + c0 + 12));
        // e_dst for the same columns (L2/L0-resident, broadcast).
        v4f d0 = *(const v4f*)(edp + m + c0);
        v4f d1 = *(const v4f*)(edp + m + c0 + 4);
        v4f d2 = *(const v4f*)(edp + m + c0 + 8);
        v4f d3 = *(const v4f*)(edp + m + c0 + 12);

        float av[16], dv[16];
#pragma unroll
        for (int j = 0; j < 4; j++) {
            av[j] = a0[j]; av[4 + j] = a1[j]; av[8 + j] = a2[j]; av[12 + j] = a3[j];
            dv[j] = d0[j]; dv[4 + j] = d1[j]; dv[8 + j] = d2[j]; dv[12 + j] = d3[j];
        }

        v16h pa;
#pragma unroll
        for (int j = 0; j < 16; j++) {
            float e = esv + dv[j];
            float t = fmaxf(e, NEG_SLOPE * e);           // LeakyReLU (monotone)
            float p = __builtin_amdgcn_exp2f(fmaf(t, C_SCALE, -bound2)); // v_exp_f32
            p *= av[j];                                  // mask: A is exactly {0,1}
            lr[j & 3] += p;
            pa[j] = (_Float16)p;
        }

        // B-operand: WxT rows (=output col), contiguous K; low lanes K 0..15,
        // high lanes K 16..31, column = t*16 + lane%16.
        const _Float16* wp = Wb + (size_t)r * NN + m + 16 * hi;
        v16h b0 = *(const v16h*)(wp);
        v16h b1 = *(const v16h*)(wp + 16 * (size_t)NN);
        v16h b2 = *(const v16h*)(wp + 32 * (size_t)NN);
        v16h b3 = *(const v16h*)(wp + 48 * (size_t)NN);

        acc0 = __builtin_amdgcn_wmma_f32_16x16x32_f16(false, pa, false, b0, (short)0, acc0, false, false);
        acc1 = __builtin_amdgcn_wmma_f32_16x16x32_f16(false, pa, false, b1, (short)0, acc1, false, false);
        acc2 = __builtin_amdgcn_wmma_f32_16x16x32_f16(false, pa, false, b2, (short)0, acc2, false, false);
        acc3 = __builtin_amdgcn_wmma_f32_16x16x32_f16(false, pa, false, b3, (short)0, acc3, false, false);
    }

    // Combine partials, then the two half-wave sums (lanes r, r+16 share a row).
    float lrun = (lr[0] + lr[1]) + (lr[2] + lr[3]);
    lrun += __shfl_xor(lrun, 16, 32);

    // Epilogue: out[b, r0+row, :] = acc / l  (l>0 guaranteed by self-loops;
    // guard matches reference's nan_to_num -> 0).
#pragma unroll
    for (int v = 0; v < 8; v++) {
        float lv  = __shfl(lrun, v + hi * 8, 32);   // C VGPR v <-> row v / v+8
        float inv = (lv > 0.f) ? 1.0f / lv : 0.f;
        int row  = r0 + v + hi * 8;
        int col  = lane & 15;
        float* op = out + ((size_t)b * NN + row) * DD + col;
        op[0]  = acc0[v] * inv;
        op[16] = acc1[v] * inv;
        op[32] = acc2[v] * inv;
        op[48] = acc3[v] * inv;
    }
}

// ---------------------------------------------------------------------------
extern "C" void kernel_launch(void* const* d_in, const int* in_sizes, int n_in,
                              void* d_out, int out_size, void* d_ws, size_t ws_size,
                              hipStream_t stream)
{
    const float* x     = (const float*)d_in[0];
    const float* A     = (const float*)d_in[1];
    const float* W     = (const float*)d_in[2];
    const float* a_src = (const float*)d_in[3];
    const float* a_dst = (const float*)d_in[4];
    float* out = (float*)d_out;

    _Float16* WxT = (_Float16*)d_ws;                                   // 2 MB
    float* es   = (float*)((char*)d_ws + (size_t)BB * DD * NN * sizeof(_Float16));
    float* ed   = es + (size_t)BB * NN;                                // 64 KB each
    float* gmax = ed + (size_t)BB * NN;                                // 32 B

    gat_stage1<<<(BB * NN) / 256, 256, 0, stream>>>(x, W, a_src, a_dst, WxT, es, ed);
    gat_gmax<<<BB, 256, 0, stream>>>(ed, gmax);
    gat_stage2<<<BB * (NN / 16), 32, 0, stream>>>(A, WxT, es, ed, gmax, out);
}
